// MLPEncoder_58514634441436
// MI455X (gfx1250) — matrix operand order repaired and checked by reading, not compiled, not run on hardware
//
#include <hip/hip_runtime.h>
#include <hip/hip_bf16.h>

#define N_NODES 50000
#define N_EDGES 800000
#define D_IN 64
#define D_HID 128
#define N_LAYERS 4
#define EPS_MSG 1e-7f
#define EPS_SOFTMAX 1e-16f
#define EPS_BN 1e-5f

typedef __attribute__((ext_vector_type(16))) _Float16 v16h;
typedef __attribute__((ext_vector_type(8)))  float    v8f;

// ---------------------------------------------------------------------------
// Zero per-layer accumulators: segment-max m, numerator, denominator, BN stats
// ---------------------------------------------------------------------------
__global__ void k_init(float* __restrict__ m, float* __restrict__ num,
                       float* __restrict__ den, float* __restrict__ stats) {
    int i = blockIdx.x * blockDim.x + threadIdx.x;
    if (i < N_NODES * D_IN) { m[i] = 0.0f; num[i] = 0.0f; den[i] = 0.0f; }
    if (i < 2 * D_HID) stats[i] = 0.0f;
}

// ---------------------------------------------------------------------------
// Pass 1 over edges: per-(dst,channel) max of msg = relu(x[src]) + eps.
// msg > 0 always, so positive-float bit patterns are monotonic -> uint atomicMax.
// 16 threads per edge, float4 per thread -> coalesced 256B row gathers.
// ---------------------------------------------------------------------------
__global__ void k_edge_max(const float* __restrict__ x, const int* __restrict__ ei,
                           unsigned int* __restrict__ m) {
    int tid = blockIdx.x * blockDim.x + threadIdx.x;
    if (tid >= N_EDGES * 16) return;
    int e  = tid >> 4;
    int c0 = (tid & 15) << 2;
    int src = ei[e];
    int dst = ei[N_EDGES + e];
    float4 xv = *(const float4*)(x + (size_t)src * D_IN + c0);
    unsigned int* md = m + (size_t)dst * D_IN + c0;
    atomicMax(md + 0, __float_as_uint(fmaxf(xv.x, 0.0f) + EPS_MSG));
    atomicMax(md + 1, __float_as_uint(fmaxf(xv.y, 0.0f) + EPS_MSG));
    atomicMax(md + 2, __float_as_uint(fmaxf(xv.z, 0.0f) + EPS_MSG));
    atomicMax(md + 3, __float_as_uint(fmaxf(xv.w, 0.0f) + EPS_MSG));
}

// ---------------------------------------------------------------------------
// Pass 2 over edges: e = exp(msg - m[dst]); den += e; num += msg*e.
// agg = num / (den + eps) later (shared denominator per (dst,channel)).
// ---------------------------------------------------------------------------
__global__ void k_edge_expsum(const float* __restrict__ x, const int* __restrict__ ei,
                              const float* __restrict__ m, float* __restrict__ num,
                              float* __restrict__ den) {
    int tid = blockIdx.x * blockDim.x + threadIdx.x;
    if (tid >= N_EDGES * 16) return;
    int e  = tid >> 4;
    int c0 = (tid & 15) << 2;
    int src = ei[e];
    int dst = ei[N_EDGES + e];
    float4 xv = *(const float4*)(x + (size_t)src * D_IN + c0);
    size_t base = (size_t)dst * D_IN + c0;
    float msg[4] = { fmaxf(xv.x, 0.0f) + EPS_MSG, fmaxf(xv.y, 0.0f) + EPS_MSG,
                     fmaxf(xv.z, 0.0f) + EPS_MSG, fmaxf(xv.w, 0.0f) + EPS_MSG };
    #pragma unroll
    for (int j = 0; j < 4; ++j) {
        float ex = __expf(msg[j] - m[base + j]);
        atomicAdd(&den[base + j], ex);
        atomicAdd(&num[base + j], msg[j] * ex);
    }
}

// ---------------------------------------------------------------------------
// GEMM1: h1[N,128] = (num/(den+eps) + x)[N,64] @ W1[64,128] + b1
// One block = 16 rows; 8 waves, each producing a 16x16 N-tile via WMMA f16.
// B is staged TRANSPOSED (Bst[n][k]) so each lane's B fragment (16 contiguous
// K values of one column, in this code's k order) is a pair of ds_load_b128 — same
// access shape as the A fragment. f32 accumulate; K=64 -> two 16x16x32 WMMAs.
// ---------------------------------------------------------------------------
__global__ void __launch_bounds__(256) k_gemm1(
        const float* __restrict__ num, const float* __restrict__ den,
        const float* __restrict__ x,   const float* __restrict__ W1,
        const float* __restrict__ b1,  float* __restrict__ h1) {
    __shared__ _Float16 As [16][D_IN];       // 2 KB   (row, k)
    __shared__ _Float16 Bst[D_HID][D_IN];    // 16 KB  (n, k)  = W1 transposed
    int tid   = threadIdx.x;
    int tileM = blockIdx.x * 16;             // 50000 = 3125*16, exact

    for (int idx = tid; idx < D_IN * D_HID; idx += 256) {
        int k = idx / D_HID, n = idx % D_HID;
        Bst[n][k] = (_Float16)W1[idx];       // coalesced global read, LDS scatter
    }
    for (int idx = tid; idx < 16 * D_IN; idx += 256) {
        int r = idx / D_IN, c = idx % D_IN;
        size_t g = (size_t)(tileM + r) * D_IN + c;
        float h = num[g] / (den[g] + EPS_SOFTMAX) + x[g];
        As[r][c] = (_Float16)h;
    }
    __syncthreads();

    int wave = tid >> 5, lane = tid & 31, lm = lane & 15, hi = lane >> 4;
    v8f acc = {};
    #pragma unroll
    for (int kb = 0; kb < 2; ++kb) {
        v16h a, b;
        #pragma unroll
        for (int h = 0; h < 16; ++h) {
            // A: lanes 0-15 hold K 0..15, lanes 16-31 K 16..31 (same as B)
            // -> contiguous 16 halves (2x ds_load_b128).
            int kl = hi * 16 + h;
            a[h] = As[lm][kb * 32 + kl];
            // B: column (wave*16+lm), halves 0..15 -> K contiguous (+16 if hi)
            // -> contiguous 16 halves in Bst row (2x ds_load_b128).
            b[h] = Bst[wave * 16 + lm][kb * 32 + hi * 16 + h];
        }
        acc = __builtin_amdgcn_wmma_f32_16x16x32_f16(
            false, a, false, b, (short)0, acc, false, false);
    }
    #pragma unroll
    for (int i = 0; i < 8; ++i) {
        int row = tileM + i + 8 * hi;        // C/D: VGPR i -> M=i / M=i+8
        int col = wave * 16 + lm;
        h1[(size_t)row * D_HID + col] = acc[i] + b1[col];
    }
}

// ---------------------------------------------------------------------------
// BN stats: per-channel sum and sum-of-squares via partial sums + atomics.
// ---------------------------------------------------------------------------
__global__ void k_bn_stats(const float* __restrict__ h1, float* __restrict__ stats) {
    int c   = threadIdx.x & (D_HID - 1);
    int sub = threadIdx.x >> 7;              // 0 or 1
    float s = 0.0f, sq = 0.0f;
    for (int r = blockIdx.x * 2 + sub; r < N_NODES; r += gridDim.x * 2) {
        float v = h1[(size_t)r * D_HID + c];
        s += v; sq += v * v;
    }
    atomicAdd(&stats[c], s);
    atomicAdd(&stats[D_HID + c], sq);
}

// ---------------------------------------------------------------------------
// BN (train-mode, biased var) + ReLU, in place on h1.
// ---------------------------------------------------------------------------
__global__ void k_bn_relu(float* __restrict__ h1, const float* __restrict__ stats,
                          const float* __restrict__ gamma, const float* __restrict__ beta) {
    int i = blockIdx.x * blockDim.x + threadIdx.x;
    if (i >= N_NODES * D_HID) return;
    int c = i & (D_HID - 1);
    const float invN = 1.0f / (float)N_NODES;
    float mean = stats[c] * invN;
    float var  = stats[D_HID + c] * invN - mean * mean;
    float rs   = rsqrtf(var + EPS_BN);
    float v    = (h1[i] - mean) * rs * gamma[c] + beta[c];
    h1[i] = fmaxf(v, 0.0f);
}

// ---------------------------------------------------------------------------
// GEMM2: out[N,64] = relu_bn(h1)[N,128] @ W2[128,64] + b2, optional Mish.
// One block = 32 rows; 8 waves = 2 M-tiles x 4 N-tiles. K=128 -> 4 WMMAs.
// B staged transposed, same as GEMM1.
// ---------------------------------------------------------------------------
__global__ void __launch_bounds__(256) k_gemm2(
        const float* __restrict__ h1, const float* __restrict__ W2,
        const float* __restrict__ b2, float* __restrict__ out, int do_mish) {
    __shared__ _Float16 As [32][D_HID];      // 8 KB   (row, k)
    __shared__ _Float16 Bst[D_IN][D_HID];    // 16 KB  (n, k)  = W2 transposed
    int tid     = threadIdx.x;
    int rowBase = blockIdx.x * 32;

    for (int idx = tid; idx < D_HID * D_IN; idx += 256) {
        int k = idx / D_IN, n = idx % D_IN;
        Bst[n][k] = (_Float16)W2[idx];
    }
    for (int idx = tid; idx < 32 * D_HID; idx += 256) {
        int r = idx >> 7, c = idx & (D_HID - 1);
        int row = rowBase + r;
        As[r][c] = (row < N_NODES) ? (_Float16)h1[(size_t)row * D_HID + c]
                                   : (_Float16)0.0f;
    }
    __syncthreads();

    int wave = tid >> 5, lane = tid & 31, lm = lane & 15, hi = lane >> 4;
    int mt = wave >> 2, nt = wave & 3;
    v8f acc = {};
    #pragma unroll
    for (int kb = 0; kb < 4; ++kb) {
        v16h a, b;
        #pragma unroll
        for (int h = 0; h < 16; ++h) {
            int kl = hi * 16 + h;
            a[h] = As[mt * 16 + lm][kb * 32 + kl];
            b[h] = Bst[nt * 16 + lm][kb * 32 + hi * 16 + h];
        }
        acc = __builtin_amdgcn_wmma_f32_16x16x32_f16(
            false, a, false, b, (short)0, acc, false, false);
    }
    #pragma unroll
    for (int i = 0; i < 8; ++i) {
        int row = rowBase + mt * 16 + i + 8 * hi;
        int col = nt * 16 + lm;
        if (row < N_NODES) {
            float v = acc[i] + b2[col];
            if (do_mish) {
                float sp = (v > 20.0f) ? v : log1pf(__expf(v));  // softplus
                v = v * tanhf(sp);                               // mish
            }
            out[(size_t)row * D_IN + col] = v;
        }
    }
}

// ---------------------------------------------------------------------------
extern "C" void kernel_launch(void* const* d_in, const int* in_sizes, int n_in,
                              void* d_out, int out_size, void* d_ws, size_t ws_size,
                              hipStream_t stream) {
    const float* x0    = (const float*)d_in[0];
    const int*   ei    = (const int*)d_in[1];   // edge_index [2, E]
    const float* W1    = (const float*)d_in[2];
    const float* b1    = (const float*)d_in[3];
    const float* gamma = (const float*)d_in[4];
    const float* beta  = (const float*)d_in[5];
    const float* W2    = (const float*)d_in[6];
    const float* b2    = (const float*)d_in[7];

    float* ws = (float*)d_ws;
    const size_t NX = (size_t)N_NODES * D_IN;
    float* m     = ws;                                  // seg max     [N,64]
    float* num   = m + NX;                              // numerator   [N,64]
    float* den   = num + NX;                            // denominator [N,64]
    float* h1    = den + NX;                            // hidden      [N,128]
    float* stats = h1 + (size_t)N_NODES * D_HID;        // sum/sumsq   [256]
    float* xbuf  = stats + 256;                         // layer io    [N,64]

    const int initGrid = (N_NODES * D_IN + 255) / 256;
    const int edgeGrid = (N_EDGES * 16 + 255) / 256;

    for (int l = 0; l < N_LAYERS; ++l) {
        const float* xin  = (l == 0) ? x0 : xbuf;
        float*       xout = (l == N_LAYERS - 1) ? (float*)d_out : xbuf;

        k_init<<<initGrid, 256, 0, stream>>>(m, num, den, stats);
        k_edge_max<<<edgeGrid, 256, 0, stream>>>(xin, ei, (unsigned int*)m);
        k_edge_expsum<<<edgeGrid, 256, 0, stream>>>(xin, ei, m, num, den);
        k_gemm1<<<N_NODES / 16, 256, 0, stream>>>(num, den, xin,
                                                  W1 + (size_t)l * D_IN * D_HID,
                                                  b1 + (size_t)l * D_HID, h1);
        k_bn_stats<<<250, 256, 0, stream>>>(h1, stats);
        k_bn_relu<<<(N_NODES * D_HID + 255) / 256, 256, 0, stream>>>(
            h1, stats, gamma + (size_t)l * D_HID, beta + (size_t)l * D_HID);
        k_gemm2<<<(N_NODES + 31) / 32, 256, 0, stream>>>(
            h1, W2 + (size_t)l * D_HID * D_IN, b2 + (size_t)l * D_IN,
            xout, (l < N_LAYERS - 1) ? 1 : 0);
    }
}
